// SimpleQuantumGNN_85873576116380
// MI455X (gfx1250) — compile-verified, matrix operand order checked by reading, not run on hardware
//
#include <hip/hip_runtime.h>
#include <hip/hip_bf16.h>
#include <stdint.h>
#include <stddef.h>

// Problem constants (match reference setup_inputs)
#define N_NODES 100000
#define N_EDGES 600000
#define IN_DIM  32
#define CDIM    128
#define NHEADS  8
#define ODIM    16
#define NGRAPH  100
#define NPG     1000
#define ETOT    (N_EDGES + N_NODES)

typedef __attribute__((ext_vector_type(16))) __bf16 v16bf;
typedef __attribute__((ext_vector_type(8)))  __bf16 v8bf;
typedef __attribute__((ext_vector_type(8)))  float  v8f;

// ---------- helpers ----------
__device__ __forceinline__ unsigned short f2bf_bits(float f) {
  unsigned u = __float_as_uint(f);
  unsigned r = u + 0x7fffu + ((u >> 16) & 1u);   // round-to-nearest-even
  return (unsigned short)(r >> 16);
}
__device__ __forceinline__ unsigned f2ord(float f) {
  unsigned u = __float_as_uint(f);
  return (u & 0x80000000u) ? ~u : (u | 0x80000000u);
}
__device__ __forceinline__ float ord2f(unsigned u) {
  unsigned v = (u & 0x80000000u) ? (u & 0x7fffffffu) : ~u;
  return __uint_as_float(v);
}
__device__ __forceinline__ float waveSum(float v) {
  #pragma unroll
  for (int m = 16; m >= 1; m >>= 1) v += __shfl_xor(v, m, 32);
  return v;
}

// ---------- generic fills / converts ----------
__global__ void k_fill_f32(float* p, float v, size_t n) {
  size_t i = (size_t)blockIdx.x * blockDim.x + threadIdx.x;
  if (i < n) p[i] = v;
}
__global__ void k_fill_u32(unsigned* p, unsigned v, size_t n) {
  size_t i = (size_t)blockIdx.x * blockDim.x + threadIdx.x;
  if (i < n) p[i] = v;
}
__global__ void k_f32_to_bf16(const float* __restrict__ x, unsigned short* __restrict__ y, size_t n) {
  size_t i = (size_t)blockIdx.x * blockDim.x + threadIdx.x;
  if (i < n) y[i] = f2bf_bits(x[i]);
}

// ---------- pack W [K,128] f32 -> B-fragment layout bf16 ----------
// Bpack index = ((ct*Kt + kt)*32 + lane)*16 + j ; value = W[k, ct*16 + (lane&15)]
// k = kt*32 + (j<8 ? kb+j : 16+kb+(j-8)), kb = (lane>>4)*8
__global__ void k_pack_B(const float* __restrict__ W, unsigned short* __restrict__ Bp, int Kt) {
  int idx = blockIdx.x * blockDim.x + threadIdx.x;
  int total = 8 * Kt * 512;
  if (idx >= total) return;
  int j    = idx & 15;
  int lane = (idx >> 4) & 31;
  int rem  = idx >> 9;
  int kt   = rem % Kt;
  int ct   = rem / Kt;
  int kb   = (lane >> 4) << 3;
  int k    = kt * 32 + ((j < 8) ? (kb + j) : (16 + kb + (j - 8)));
  int n    = ct * 16 + (lane & 15);
  Bp[idx] = f2bf_bits(W[(size_t)k * CDIM + n]);
}

// ---------- WMMA GEMM: Y[N,128] = A[N,K](bf16) @ Wpacked(bf16), f32 accum ----------
// grid.x = N/16 row tiles, 256 threads = 8 waves, wave w -> column tile w (16 cols)
__global__ void __launch_bounds__(256) k_gemm_bf16(
    const unsigned short* __restrict__ Abits,
    const unsigned short* __restrict__ Bbits,
    float* __restrict__ Y, int K) {
  const __bf16* A  = reinterpret_cast<const __bf16*>(Abits);
  const __bf16* Bp = reinterpret_cast<const __bf16*>(Bbits);
  int tile = blockIdx.x;
  int wave = threadIdx.x >> 5;
  int lane = threadIdx.x & 31;
  int Kt = K >> 5;
  int m  = lane & 15;
  int kb = (lane >> 4) << 3;
  const __bf16* arow = A + (size_t)(tile * 16 + m) * K;
  v8f acc = {0.f, 0.f, 0.f, 0.f, 0.f, 0.f, 0.f, 0.f};
  for (int kt = 0; kt < Kt; ++kt) {
    v8bf lo = *(const v8bf*)(arow + kt * 32 + kb);        // K = kt*32 + kb + 0..7
    v8bf hi = *(const v8bf*)(arow + kt * 32 + 16 + kb);   // K = kt*32 + 16 + kb + 0..7
    v16bf a = __builtin_shufflevector(lo, hi, 0,1,2,3,4,5,6,7,8,9,10,11,12,13,14,15);
    v16bf b = *(const v16bf*)(Bp + ((size_t)(wave * Kt + kt) * 32 + lane) * 16);
    acc = __builtin_amdgcn_wmma_f32_16x16x32_bf16(false, a, false, b, (short)0, acc, false, false);
  }
  int rowbase = tile * 16 + ((lane >> 4) << 3);
  int col = wave * 16 + (lane & 15);
  #pragma unroll
  for (int r = 0; r < 8; ++r)
    Y[(size_t)(rowbase + r) * CDIM + col] = acc[r];
}

// ---------- fused bias + LayerNorm + ReLU (+ bf16 copy), one wave per node ----------
__global__ void __launch_bounds__(256) k_bias_ln_relu(
    const float* __restrict__ X, const float* __restrict__ bias,
    const float* __restrict__ g, const float* __restrict__ bt,
    float* __restrict__ Ho, unsigned short* __restrict__ Hbf, int N) {
  int node = blockIdx.x * 8 + (threadIdx.x >> 5);
  int lane = threadIdx.x & 31;
  if (node >= N) return;
  const float* row = X + (size_t)node * CDIM;
  float v[4]; float s = 0.f, q = 0.f;
  #pragma unroll
  for (int t = 0; t < 4; ++t) {
    int c = lane * 4 + t;
    v[t] = row[c] + bias[c];
    s += v[t]; q += v[t] * v[t];
  }
  s = waveSum(s); q = waveSum(q);
  float mu = s * (1.f / 128.f);
  float var = q * (1.f / 128.f) - mu * mu;
  float rs = rsqrtf(var + 1e-5f);
  #pragma unroll
  for (int t = 0; t < 4; ++t) {
    int c = lane * 4 + t;
    float y = g[c] * (v[t] - mu) * rs + bt[c];
    y = fmaxf(y, 0.f);
    Ho[(size_t)node * CDIM + c] = y;
    Hbf[(size_t)node * CDIM + c] = f2bf_bits(y);
  }
}

// ---------- per-node attention projections: (xh * a).sum over O per head ----------
__global__ void __launch_bounds__(256) k_attn_node(
    const float* __restrict__ XH, const float* __restrict__ asrc, const float* __restrict__ adst,
    float* __restrict__ As, float* __restrict__ Ad, int N) {
  int node = blockIdx.x * 8 + (threadIdx.x >> 5);
  int lane = threadIdx.x & 31;
  if (node >= N) return;
  const float* row = XH + (size_t)node * CDIM;
  int head = lane >> 2;
  float ss = 0.f, sd = 0.f;
  #pragma unroll
  for (int t = 0; t < 4; ++t) {
    int c = lane * 4 + t;           // column within 128
    int o = (lane & 3) * 4 + t;     // offset within head
    float xv = row[c];
    ss += xv * asrc[head * ODIM + o];
    sd += xv * adst[head * ODIM + o];
  }
  ss += __shfl_xor(ss, 1, 32); ss += __shfl_xor(ss, 2, 32);
  sd += __shfl_xor(sd, 1, 32); sd += __shfl_xor(sd, 2, 32);
  if ((lane & 3) == 0) {
    As[(size_t)node * NHEADS + head] = ss;
    Ad[(size_t)node * NHEADS + head] = sd;
  }
}

// ---------- edge_attr column sums (for mean over E) ----------
__global__ void k_ea_colsum(const float* __restrict__ ea, float* acc, int E) {
  float s[8] = {0,0,0,0,0,0,0,0};
  for (int e = blockIdx.x * blockDim.x + threadIdx.x; e < E; e += gridDim.x * blockDim.x) {
    #pragma unroll
    for (int d = 0; d < 8; ++d) s[d] += ea[(size_t)e * 8 + d];
  }
  #pragma unroll
  for (int d = 0; d < 8; ++d) atomicAdd(&acc[d], s[d]);
}

// ---------- M[d][h] = sum_o ew[d, h*16+o] * ae[h,o]  (8x8) ----------
__global__ void k_escoreM(const float* __restrict__ ew, const float* __restrict__ ae, float* M) {
  int idx = threadIdx.x;
  if (idx >= 64) return;
  int d = idx >> 3, h = idx & 7;
  float s = 0.f;
  #pragma unroll
  for (int o = 0; o < ODIM; ++o) s += ew[(size_t)d * CDIM + h * ODIM + o] * ae[h * ODIM + o];
  M[d * 8 + h] = s;
}

// ---------- self-loop edge score: sc[h] = sum_d mean_ea[d]*M[d][h] ----------
__global__ void k_sc(const float* __restrict__ acc, const float* __restrict__ M, float* sc, float invE) {
  int h = threadIdx.x;
  if (h >= 8) return;
  float s = 0.f;
  #pragma unroll
  for (int d = 0; d < 8; ++d) s += (acc[d] * invE) * M[d * 8 + h];
  sc[h] = s;
}

// ---------- GAT pass 1: scores + segment max (ordered-uint atomicMax) ----------
__global__ void k_gat_scores(
    const int* __restrict__ src, const int* __restrict__ dst, const float* __restrict__ ea,
    const float* __restrict__ M, const float* __restrict__ sc,
    const float* __restrict__ As, const float* __restrict__ Ad,
    float* __restrict__ Aout, unsigned* __restrict__ mOrd) {
  int e = blockIdx.x * blockDim.x + threadIdx.x;
  if (e >= ETOT) return;
  int s, d;
  float esc[8];
  if (e < N_EDGES) {
    s = src[e]; d = dst[e];
    float av[8];
    #pragma unroll
    for (int dd = 0; dd < 8; ++dd) av[dd] = ea[(size_t)e * 8 + dd];
    #pragma unroll
    for (int h = 0; h < 8; ++h) {
      float t = 0.f;
      #pragma unroll
      for (int dd = 0; dd < 8; ++dd) t += av[dd] * M[dd * 8 + h];
      esc[h] = t;
    }
  } else {
    s = d = e - N_EDGES;
    #pragma unroll
    for (int h = 0; h < 8; ++h) esc[h] = sc[h];
  }
  #pragma unroll
  for (int h = 0; h < 8; ++h) {
    float a = As[(size_t)s * NHEADS + h] + Ad[(size_t)d * NHEADS + h] + esc[h];
    a = (a > 0.f) ? a : 0.2f * a;           // leaky_relu 0.2
    Aout[(size_t)e * NHEADS + h] = a;
    atomicMax(&mOrd[(size_t)d * NHEADS + h], f2ord(a));
  }
}

// ---------- GAT pass 2: exp(a - m[dst]) and denominator ----------
__global__ void k_gat_exp(
    const int* __restrict__ dst, float* __restrict__ Aout,
    const unsigned* __restrict__ mOrd, float* __restrict__ den) {
  int e = blockIdx.x * blockDim.x + threadIdx.x;
  if (e >= ETOT) return;
  int d = (e < N_EDGES) ? dst[e] : (e - N_EDGES);
  #pragma unroll
  for (int h = 0; h < 8; ++h) {
    float ex = expf(Aout[(size_t)e * NHEADS + h] - ord2f(mOrd[(size_t)d * NHEADS + h]));
    Aout[(size_t)e * NHEADS + h] = ex;
    atomicAdd(&den[(size_t)d * NHEADS + h], ex);
  }
}

// ---------- GAT pass 3: scatter alpha * xh[src] into out[dst] ----------
__global__ void k_gat_scatter(
    const int* __restrict__ src, const int* __restrict__ dst,
    const float* __restrict__ Aout, const float* __restrict__ den,
    const float* __restrict__ XH, float* __restrict__ OUT) {
  long long idx = (long long)blockIdx.x * blockDim.x + threadIdx.x;
  if (idx >= (long long)ETOT * NHEADS) return;
  int e = (int)(idx >> 3);
  int h = (int)(idx & 7);
  int s, d;
  if (e < N_EDGES) { s = src[e]; d = dst[e]; } else { s = d = e - N_EDGES; }
  float alpha = Aout[(size_t)e * NHEADS + h] / (den[(size_t)d * NHEADS + h] + 1e-16f);
  const float* xr = XH + (size_t)s * CDIM + h * ODIM;
  float* orow = OUT + (size_t)d * CDIM + h * ODIM;
  #pragma unroll
  for (int o = 0; o < ODIM; ++o) atomicAdd(&orow[o], alpha * xr[o]);
}

// ---------- GCN degree / normalization / scatter ----------
__global__ void k_deg(const int* __restrict__ dst, float* __restrict__ deg) {
  int e = blockIdx.x * blockDim.x + threadIdx.x;
  if (e < N_EDGES) atomicAdd(&deg[dst[e]], 1.0f);
}
__global__ void k_dinv(float* __restrict__ deg) {
  int n = blockIdx.x * blockDim.x + threadIdx.x;
  if (n < N_NODES) deg[n] = rsqrtf(deg[n]);   // deg >= 1 always (self-loop)
}
__global__ void k_gcn_scatter(
    const int* __restrict__ src, const int* __restrict__ dst,
    const float* __restrict__ dinv, const float* __restrict__ XH, float* __restrict__ OUT) {
  long long idx = (long long)blockIdx.x * blockDim.x + threadIdx.x;
  if (idx >= (long long)ETOT * NHEADS) return;
  int e = (int)(idx >> 3);
  int h = (int)(idx & 7);
  int s, d;
  if (e < N_EDGES) { s = src[e]; d = dst[e]; } else { s = d = e - N_EDGES; }
  float nrm = dinv[s] * dinv[d];
  const float* xr = XH + (size_t)s * CDIM + h * ODIM;
  float* orow = OUT + (size_t)d * CDIM + h * ODIM;
  #pragma unroll
  for (int o = 0; o < ODIM; ++o) atomicAdd(&orow[o], nrm * xr[o]);
}

// ---------- readout: mean / max / sum per contiguous graph ----------
__global__ void __launch_bounds__(128) k_pool(const float* __restrict__ H, float* __restrict__ P) {
  int g = blockIdx.x, c = threadIdx.x;
  float s = 0.f, mx = -3.402823466e38f;
  const float* base = H + (size_t)g * NPG * CDIM + c;
  for (int n = 0; n < NPG; ++n) {
    float v = base[(size_t)n * CDIM];
    s += v; mx = fmaxf(mx, v);
  }
  P[(size_t)g * 384 + c]       = s * (1.f / NPG);
  P[(size_t)g * 384 + 128 + c] = mx;
  P[(size_t)g * 384 + 256 + c] = s;
}

// ---------- MLP layer with fused LN+ReLU (128 outputs, 100 rows) ----------
__global__ void __launch_bounds__(128) k_mlp_ln(
    const float* __restrict__ X, const float* __restrict__ W, const float* __restrict__ b,
    const float* __restrict__ g, const float* __restrict__ bt, float* __restrict__ Z, int K) {
  int gr = blockIdx.x, c = threadIdx.x;
  const float* xr = X + (size_t)gr * K;
  float acc = b[c];
  for (int k = 0; k < K; ++k) acc += xr[k] * W[(size_t)k * CDIM + c];
  __shared__ float shs[4], shq[4];
  float s = acc, q = acc * acc;
  #pragma unroll
  for (int m = 16; m >= 1; m >>= 1) { s += __shfl_xor(s, m, 32); q += __shfl_xor(q, m, 32); }
  if ((threadIdx.x & 31) == 0) { shs[threadIdx.x >> 5] = s; shq[threadIdx.x >> 5] = q; }
  __syncthreads();
  float ts = shs[0] + shs[1] + shs[2] + shs[3];
  float tq = shq[0] + shq[1] + shq[2] + shq[3];
  float mu = ts * (1.f / 128.f);
  float var = tq * (1.f / 128.f) - mu * mu;
  float rs = rsqrtf(var + 1e-5f);
  float y = g[c] * (acc - mu) * rs + bt[c];
  Z[(size_t)gr * CDIM + c] = fmaxf(y, 0.f);
}

__global__ void k_mlp_out(const float* __restrict__ Z, const float* __restrict__ w,
                          const float* __restrict__ b, float* __restrict__ out, int G) {
  int g = blockIdx.x * blockDim.x + threadIdx.x;
  if (g >= G) return;
  float acc = b[0];
  for (int k = 0; k < CDIM; ++k) acc += Z[(size_t)g * CDIM + k] * w[k];
  out[g] = acc;
}

// =====================================================================
extern "C" void kernel_launch(void* const* d_in, const int* in_sizes, int n_in,
                              void* d_out, int out_size, void* d_ws, size_t ws_size,
                              hipStream_t stream) {
  (void)in_sizes; (void)n_in; (void)out_size; (void)ws_size;

  const float* x   = (const float*)d_in[0];
  const int*   ei  = (const int*)d_in[1];
  const float* ea  = (const float*)d_in[2];
  const int* src = ei;
  const int* dst = ei + N_EDGES;

  const float* in_w    = (const float*)d_in[4];
  const float* in_b    = (const float*)d_in[5];
  const float* in_ln_g = (const float*)d_in[6];
  const float* in_ln_b = (const float*)d_in[7];
  // param base index per layer: GAT layers have 8 params, GCN layers 4
  const int base[5] = {8, 16, 20, 28, 32};

  // workspace carve-out
  char* wsb = (char*)d_ws;
  size_t off = 0;
  auto alloc = [&](size_t bytes) -> char* {
    char* p = wsb + off;
    off += (bytes + 255) & ~(size_t)255;
    return p;
  };
  float*          bufA    = (float*)alloc((size_t)N_NODES * CDIM * 4);   // h / aggregate target
  float*          bufB    = (float*)alloc((size_t)N_NODES * CDIM * 4);   // xh (GEMM out)
  unsigned short* hbf     = (unsigned short*)alloc((size_t)N_NODES * CDIM * 2);
  float*          aEx     = (float*)alloc((size_t)ETOT * NHEADS * 4);
  unsigned*       mOrd    = (unsigned*)alloc((size_t)N_NODES * NHEADS * 4);
  float*          den     = (float*)alloc((size_t)N_NODES * NHEADS * 4);
  float*          asrcN   = (float*)alloc((size_t)N_NODES * NHEADS * 4);
  float*          adstN   = (float*)alloc((size_t)N_NODES * NHEADS * 4);
  float*          deg     = (float*)alloc((size_t)N_NODES * 4);
  unsigned short* Bpack   = (unsigned short*)alloc((size_t)CDIM * CDIM * 2);
  float*          meanAcc = (float*)alloc(8 * 4);
  float*          Mbuf    = (float*)alloc(64 * 4);
  float*          scBuf   = (float*)alloc(8 * 4);
  float*          pooled  = (float*)alloc((size_t)NGRAPH * 3 * CDIM * 4);
  float*          z0      = (float*)alloc((size_t)NGRAPH * CDIM * 4);
  float*          z1      = (float*)alloc((size_t)NGRAPH * CDIM * 4);

  const int rowTiles = N_NODES / 16;                 // 6250 (exact)
  const int nodeBlks = N_NODES / 8;                  // 12500 (one wave per node)
  const int eBlks    = (ETOT + 255) / 256;
  const int ehBlks   = (int)(((long long)ETOT * NHEADS + 255) / 256);

  // ---- input projection: x(bf16) @ in_w -> bufB, then LN+ReLU -> bufA + hbf ----
  {
    size_t n = (size_t)N_NODES * IN_DIM;
    k_f32_to_bf16<<<(unsigned)((n + 255) / 256), 256, 0, stream>>>(x, hbf, n);
    k_pack_B<<<16, 256, 0, stream>>>(in_w, Bpack, 1);
    k_gemm_bf16<<<rowTiles, 256, 0, stream>>>(hbf, Bpack, bufB, IN_DIM);
    k_bias_ln_relu<<<nodeBlks, 256, 0, stream>>>(bufB, in_b, in_ln_g, in_ln_b, bufA, hbf, N_NODES);
  }

  // ---- mean(edge_attr) column sums (done once per call) ----
  k_fill_f32<<<1, 32, 0, stream>>>(meanAcc, 0.f, 8);
  k_ea_colsum<<<256, 256, 0, stream>>>(ea, meanAcc, N_EDGES);

  // ---- 5 layers: GAT (0,2,4) / GCN (1,3) ----
  for (int i = 0; i < 5; ++i) {
    int b = base[i];
    const float* W = (const float*)d_in[b];
    // projection xh = h @ W via WMMA
    k_pack_B<<<64, 256, 0, stream>>>(W, Bpack, 4);
    k_gemm_bf16<<<rowTiles, 256, 0, stream>>>(hbf, Bpack, bufB, CDIM);

    if ((i & 1) == 0) {  // GAT
      const float* p_asrc = (const float*)d_in[b + 1];
      const float* p_adst = (const float*)d_in[b + 2];
      const float* p_ew   = (const float*)d_in[b + 3];
      const float* p_ae   = (const float*)d_in[b + 4];
      const float* p_b    = (const float*)d_in[b + 5];
      const float* p_g    = (const float*)d_in[b + 6];
      const float* p_bt   = (const float*)d_in[b + 7];

      k_attn_node<<<nodeBlks, 256, 0, stream>>>(bufB, p_asrc, p_adst, asrcN, adstN, N_NODES);
      k_escoreM<<<1, 64, 0, stream>>>(p_ew, p_ae, Mbuf);
      k_sc<<<1, 8, 0, stream>>>(meanAcc, Mbuf, scBuf, 1.0f / (float)N_EDGES);

      size_t nh = (size_t)N_NODES * NHEADS;
      k_fill_u32<<<(unsigned)((nh + 255) / 256), 256, 0, stream>>>(mOrd, 0u, nh);
      k_fill_f32<<<(unsigned)((nh + 255) / 256), 256, 0, stream>>>(den, 0.f, nh);
      size_t nc = (size_t)N_NODES * CDIM;
      k_fill_f32<<<(unsigned)((nc + 255) / 256), 256, 0, stream>>>(bufA, 0.f, nc);

      k_gat_scores<<<eBlks, 256, 0, stream>>>(src, dst, ea, Mbuf, scBuf, asrcN, adstN, aEx, mOrd);
      k_gat_exp<<<eBlks, 256, 0, stream>>>(dst, aEx, mOrd, den);
      k_gat_scatter<<<ehBlks, 256, 0, stream>>>(src, dst, aEx, den, bufB, bufA);

      k_bias_ln_relu<<<nodeBlks, 256, 0, stream>>>(bufA, p_b, p_g, p_bt, bufA, hbf, N_NODES);
    } else {             // GCN
      const float* p_b  = (const float*)d_in[b + 1];
      const float* p_g  = (const float*)d_in[b + 2];
      const float* p_bt = (const float*)d_in[b + 3];

      k_fill_f32<<<(N_NODES + 255) / 256, 256, 0, stream>>>(deg, 1.0f, (size_t)N_NODES);
      k_deg<<<(N_EDGES + 255) / 256, 256, 0, stream>>>(dst, deg);
      k_dinv<<<(N_NODES + 255) / 256, 256, 0, stream>>>(deg);
      size_t nc = (size_t)N_NODES * CDIM;
      k_fill_f32<<<(unsigned)((nc + 255) / 256), 256, 0, stream>>>(bufA, 0.f, nc);
      k_gcn_scatter<<<ehBlks, 256, 0, stream>>>(src, dst, deg, bufB, bufA);

      k_bias_ln_relu<<<nodeBlks, 256, 0, stream>>>(bufA, p_b, p_g, p_bt, bufA, hbf, N_NODES);
    }
  }

  // ---- readout + MLP head ----
  k_pool<<<NGRAPH, 128, 0, stream>>>(bufA, pooled);
  k_mlp_ln<<<NGRAPH, 128, 0, stream>>>(pooled,
      (const float*)d_in[40], (const float*)d_in[41],
      (const float*)d_in[42], (const float*)d_in[43], z0, 3 * CDIM);
  k_mlp_ln<<<NGRAPH, 128, 0, stream>>>(z0,
      (const float*)d_in[44], (const float*)d_in[45],
      (const float*)d_in[46], (const float*)d_in[47], z1, CDIM);
  k_mlp_out<<<1, 128, 0, stream>>>(z1,
      (const float*)d_in[48], (const float*)d_in[49], (float*)d_out, NGRAPH);
}
